// MultiHeadAdditiveAttention_16647293239357
// MI455X (gfx1250) — matrix operand order repaired and checked by reading, not compiled, not run on hardware
//
#include <hip/hip_runtime.h>
#include <hip/hip_bf16.h>
#include <math.h>

// ---------------------------------------------------------------------------
// MultiHeadAdditiveAttention for MI455X (gfx1250, wave32, WMMA)
// Dominant cost: two [32768x1024]x[1024x1024] GEMMs (137 GFLOP) -> bf16 WMMA.
// Global->LDS staging uses CDNA5 async copies (GLOBAL_LOAD_ASYNC_TO_LDS_B128,
// ASYNCcnt) so no VGPRs are burned on prefetch staging (fixes scratch spills
// seen in the previous round).
// ---------------------------------------------------------------------------

typedef __bf16 v16bf __attribute__((ext_vector_type(16)));
typedef float  v8f   __attribute__((ext_vector_type(8)));

static constexpr int Bb = 8, Ll = 4096, Dd = 1024, Hh = 16, DKk = 64;
static constexpr int Mm = Bb * Ll;          // 32768 rows

union Frag32 {           // one 16x32 (A) or 32x16 (B) bf16 fragment: 16 bf16/lane
  v16bf v;
  uint4 q[2];
};

__device__ __forceinline__ v8f wmma_bf16(v16bf a, v16bf b, v8f c) {
  // D = A(16x32) * B(32x16) + C(16x16 f32)
  return __builtin_amdgcn_wmma_f32_16x16x32_bf16(false, a, false, b,
                                                 (short)0, c, false, false);
}

// CDNA5 async copy: 16B per lane, global -> LDS, tracked by ASYNCcnt.
// ldsAddr = low 32 bits of a generic pointer to __shared__ (== LDS byte offset).
__device__ __forceinline__ void async_b128(unsigned ldsAddr, const void* gaddr) {
  asm volatile("global_load_async_to_lds_b128 %0, %1, off"
               :: "v"(ldsAddr), "v"(gaddr) : "memory");
}
__device__ __forceinline__ void wait_asynccnt_le4() {
  asm volatile("s_wait_asynccnt 4" ::: "memory");
}

// ---------------- prep: fp32 -> bf16 conversions / transposes ---------------

__global__ void k_cvt(const float* __restrict__ src, __bf16* __restrict__ dst) {
  const int i = blockIdx.x * 256 + threadIdx.x;          // 4 elems / thread
  float4 v = ((const float4*)src)[i];
  union { __bf16 h[4]; uint2 u; } p;
  p.h[0] = (__bf16)v.x; p.h[1] = (__bf16)v.y;
  p.h[2] = (__bf16)v.z; p.h[3] = (__bf16)v.w;
  ((uint2*)dst)[i] = p.u;
}

// W [k][n] (1024x1024 f32, row-major) -> WT [n][k] bf16
__global__ void k_trans(const float* __restrict__ src, __bf16* __restrict__ dst) {
  const int idx = blockIdx.x * 256 + threadIdx.x;        // idx = n*1024 + k
  const int k = idx & 1023, n = idx >> 10;
  dst[idx] = (__bf16)src[k * 1024 + n];
}

// Wv [k][16] -> WvT [h][k] bf16
__global__ void k_trans_wv(const float* __restrict__ src, __bf16* __restrict__ dst) {
  const int idx = blockIdx.x * 256 + threadIdx.x;        // idx = h*1024 + k
  const int k = idx & 1023, h = idx >> 10;
  dst[idx] = (__bf16)src[k * Hh + h];
}

// -------- k_gemm1: s16 = tanh(q@Wq + m@Wk + bq + bk)  [bf16 out] ------------
// Tile: 128(M) x 128(N) x 32(K) per WG, K total = 2048 (query then memory).
// 8 waves in 4(m) x 2(n); each wave: 32x64 output = 2x4 WMMA accumulators.
// 3 LDS buffers, async prefetch distance 2, one barrier + one async-wait/step.
// LDS rows padded to 80B (40 bf16) -> conflict-free ds_load_b128 frag reads.

__global__ __launch_bounds__(256) void k_gemm1(
    const __bf16* __restrict__ qb, const __bf16* __restrict__ mb,
    const __bf16* __restrict__ WqT, const __bf16* __restrict__ WkT,
    const float* __restrict__ bq, const float* __restrict__ bk,
    __bf16* __restrict__ s16)
{
  __shared__ __align__(16) unsigned short As[3][128 * 40];   // 3 x 10 KB
  __shared__ __align__(16) unsigned short Bs[3][128 * 40];   // 3 x 10 KB

  const int t     = threadIdx.x;
  const int lane  = t & 31;
  const int wid   = t >> 5;
  const int waveM = wid >> 1;               // 0..3 -> rows waveM*32
  const int waveN = wid & 1;                // 0..1 -> cols waveN*64
  const int lm    = lane & 15;
  const int hi    = lane >> 4;
  const int mBlk  = blockIdx.y * 128;
  const int nBlk  = blockIdx.x * 128;
  const int r0    = t >> 2;                 // fill rows r0 and r0+64
  const int c0    = t & 3;                  // fill 16B chunk within row

  // per-thread element offsets for the async fills (K offset added per step)
  const size_t aOff0 = (size_t)(mBlk + r0)      * Dd + c0 * 8;
  const size_t aOff1 = (size_t)(mBlk + r0 + 64) * Dd + c0 * 8;
  const size_t bOff0 = (size_t)(nBlk + r0)      * Dd + c0 * 8;
  const size_t bOff1 = (size_t)(nBlk + r0 + 64) * Dd + c0 * 8;

  // LDS destination byte offsets (buffer 0); buffer stride = 10240 bytes
  const unsigned ldsStride = 128 * 40 * 2;
  const unsigned ldsA0 = (unsigned)(size_t)&As[0][(r0)      * 40 + c0 * 8];
  const unsigned ldsA1 = (unsigned)(size_t)&As[0][(r0 + 64) * 40 + c0 * 8];
  const unsigned ldsB0 = (unsigned)(size_t)&Bs[0][(r0)      * 40 + c0 * 8];
  const unsigned ldsB1 = (unsigned)(size_t)&Bs[0][(r0 + 64) * 40 + c0 * 8];

  auto issue_group = [&](int ks, int buf) {
    const int kk = ks * 32;
    const __bf16* Asrc = (kk < Dd) ? qb  : mb;
    const __bf16* Bsrc = (kk < Dd) ? WqT : WkT;
    const size_t kq = (size_t)(kk & (Dd - 1));
    const unsigned bo = (unsigned)buf * ldsStride;
    async_b128(ldsA0 + bo, Asrc + aOff0 + kq);
    async_b128(ldsA1 + bo, Asrc + aOff1 + kq);
    async_b128(ldsB0 + bo, Bsrc + bOff0 + kq);
    async_b128(ldsB1 + bo, Bsrc + bOff1 + kq);
  };

  v8f acc[2][4] = {};

  issue_group(0, 0);
  issue_group(1, 1);
  wait_asynccnt_le4();                      // group 0 landed in LDS
  __syncthreads();

  int cur = 0;
  for (int ks = 0; ks < 64; ++ks) {
    // ---- compute on buffer `cur` (group ks) ----
    Frag32 af0, af1;
    {
      const int m0 = waveM * 32 + lm;
      af0.q[0] = *(const uint4*)&As[cur][m0 * 40 + hi * 8];
      af0.q[1] = *(const uint4*)&As[cur][m0 * 40 + 16 + hi * 8];
      af1.q[0] = *(const uint4*)&As[cur][(m0 + 16) * 40 + hi * 8];
      af1.q[1] = *(const uint4*)&As[cur][(m0 + 16) * 40 + 16 + hi * 8];
    }
#pragma unroll
    for (int j = 0; j < 4; ++j) {           // one B frag live at a time
      const int n = waveN * 64 + j * 16 + lm;
      Frag32 bf;
      bf.q[0] = *(const uint4*)&Bs[cur][n * 40 + hi * 8];
      bf.q[1] = *(const uint4*)&Bs[cur][n * 40 + 16 + hi * 8];
      acc[0][j] = wmma_bf16(af0.v, bf.v, acc[0][j]);
      acc[1][j] = wmma_bf16(af1.v, bf.v, acc[1][j]);
    }

    // ---- prefetch group ks+2 into buffer (cur+2)%3 == (cur-1)%3 ----
    // That buffer was last read at iteration ks-1; every wave passed the
    // barrier at the end of iteration ks-1, so the overwrite is safe.
    if (ks + 2 < 64) issue_group(ks + 2, (cur == 0) ? 2 : cur - 1);

    wait_asynccnt_le4();                    // group ks+1 complete (only ks+2 left)
    __syncthreads();                        // make it visible to all waves
    cur = (cur == 2) ? 0 : cur + 1;
  }

  // epilogue: + (bq+bk), tanh, store bf16
#pragma unroll
  for (int j = 0; j < 4; ++j) {
    const int gn = nBlk + waveN * 64 + j * 16 + lm;
    const float bsum = bq[gn] + bk[gn];
#pragma unroll
    for (int i = 0; i < 2; ++i) {
      const int gm0 = mBlk + waveM * 32 + i * 16 + hi * 8;
#pragma unroll
      for (int r = 0; r < 8; ++r) {
        const float v = tanhf(acc[i][j][r] + bsum);
        s16[(size_t)(gm0 + r) * Dd + gn] = (__bf16)v;
      }
    }
  }
}

// ---------- k_gemm2: logits[b*16+h][l] = s16 @ Wv + bv (WMMA, N=H=16) -------

__global__ __launch_bounds__(256) void k_gemm2(
    const __bf16* __restrict__ s16, const __bf16* __restrict__ WvT,
    const float* __restrict__ bv, float* __restrict__ logits)
{
  const int t = threadIdx.x, lane = t & 31, wid = t >> 5;
  const int tile = blockIdx.x * 8 + wid;    // 2048 tiles of 16 rows
  const int lm = lane & 15, hi = lane >> 4;
  const int m0 = tile * 16;

  v8f acc = {};
  const __bf16* arow = s16 + (size_t)(m0 + lm) * Dd;
  const __bf16* brow = WvT + (size_t)lm * Dd;
#pragma unroll 4
  for (int k = 0; k < Dd; k += 32) {
    Frag32 a, b;
    a.q[0] = *(const uint4*)(arow + k + hi * 8);
    a.q[1] = *(const uint4*)(arow + k + 16 + hi * 8);
    b.q[0] = *(const uint4*)(brow + k + hi * 8);
    b.q[1] = *(const uint4*)(brow + k + 16 + hi * 8);
    acc = wmma_bf16(a.v, b.v, acc);
  }
  const float bvv = bv[lm];
#pragma unroll
  for (int r = 0; r < 8; ++r) {
    const int m = m0 + r + hi * 8;
    const int b = m >> 12, l = m & (Ll - 1);
    logits[((size_t)(b * Hh + lm)) * Ll + l] = acc[r] + bvv;
  }
}

// -------- k_softmax: softmax over L + weighted sum of memory head-slice -----

__global__ __launch_bounds__(256) void k_softmax(
    const float* __restrict__ logits, const float* __restrict__ bias,
    const float* __restrict__ memory, float* __restrict__ xbuf)
{
  __shared__ float w[Ll];
  __shared__ float red[256];
  const int bh = blockIdx.x;
  const int b = bh >> 4, h = bh & 15;
  const int t = threadIdx.x;
  const float* lg = logits + (size_t)bh * Ll;
  const float* bs = bias + (size_t)b * Ll;

  float mx = -3.402823e38f;
  for (int l = t; l < Ll; l += 256) mx = fmaxf(mx, lg[l] + bs[l]);
  red[t] = mx; __syncthreads();
  for (int s = 128; s > 0; s >>= 1) {
    if (t < s) red[t] = fmaxf(red[t], red[t + s]);
    __syncthreads();
  }
  const float smax = red[0]; __syncthreads();

  float sum = 0.f;
  for (int l = t; l < Ll; l += 256) {
    const float e = __expf(lg[l] + bs[l] - smax);
    w[l] = e; sum += e;
  }
  red[t] = sum; __syncthreads();
  for (int s = 128; s > 0; s >>= 1) {
    if (t < s) red[t] += red[t + s];
    __syncthreads();
  }
  const float inv = 1.0f / red[0]; __syncthreads();

  const int dk = t & 63, lgp = t >> 6;      // 64 dk-lanes x 4 l-groups
  float acc = 0.f;
  const float* mem = memory + (size_t)b * Ll * Dd + h * DKk + dk;
  for (int l = lgp; l < Ll; l += 4) acc += w[l] * mem[(size_t)l * Dd];
  red[t] = acc; __syncthreads();
  if (t < 64)
    xbuf[(size_t)bh * DKk + t] =
        (red[t] + red[t + 64] + red[t + 128] + red[t + 192]) * inv;
}

// ----------------- k_out: out = x @ Wo + bo  (8x1024 x 1024x1024) -----------

__global__ __launch_bounds__(256) void k_out(
    const float* __restrict__ xbuf, const float* __restrict__ Wo,
    const float* __restrict__ bo, float* __restrict__ out)
{
  __shared__ float xs[Dd];
  const int b = blockIdx.y;
  const int n = blockIdx.x * 256 + threadIdx.x;
  for (int d = threadIdx.x; d < Dd; d += 256) xs[d] = xbuf[b * Dd + d];
  __syncthreads();
  float acc = bo[n];
  for (int d = 0; d < Dd; ++d) acc += xs[d] * Wo[(size_t)d * Dd + n];
  out[b * Dd + n] = acc;
}

// ---------------------------------------------------------------------------

extern "C" void kernel_launch(void* const* d_in, const int* in_sizes, int n_in,
                              void* d_out, int out_size, void* d_ws, size_t ws_size,
                              hipStream_t stream)
{
  const float* query  = (const float*)d_in[0];
  const float* bias   = (const float*)d_in[1];
  const float* memory = (const float*)d_in[2];
  const float* Wq = (const float*)d_in[3];
  const float* bq = (const float*)d_in[4];
  const float* Wk = (const float*)d_in[5];
  const float* bk = (const float*)d_in[6];
  const float* Wv = (const float*)d_in[7];
  const float* bv = (const float*)d_in[8];
  const float* Wo = (const float*)d_in[9];
  const float* bo = (const float*)d_in[10];
  float* out = (float*)d_out;

  char* ws = (char*)d_ws;
  size_t off = 0;
  __bf16* qb16 = (__bf16*)(ws + off); off += (size_t)Mm * Dd * 2;   // 64 MB
  __bf16* mb16 = (__bf16*)(ws + off); off += (size_t)Mm * Dd * 2;   // 64 MB
  __bf16* WqT  = (__bf16*)(ws + off); off += (size_t)Dd * Dd * 2;   //  2 MB
  __bf16* WkT  = (__bf16*)(ws + off); off += (size_t)Dd * Dd * 2;   //  2 MB
  __bf16* WvT  = (__bf16*)(ws + off); off += (size_t)Hh * Dd * 2;   // 32 KB
  __bf16* s16  = (__bf16*)(ws + off); off += (size_t)Mm * Dd * 2;   // 64 MB
  float*  logits = (float*)(ws + off); off += (size_t)Bb * Hh * Ll * 4; // 2 MB
  float*  xbuf   = (float*)(ws + off); off += (size_t)Bb * Hh * DKk * 4;

  // prep: bf16 copies + transposed weights
  k_cvt<<<(Mm * Dd) / 1024, 256, 0, stream>>>(query,  qb16);
  k_cvt<<<(Mm * Dd) / 1024, 256, 0, stream>>>(memory, mb16);
  k_trans<<<(Dd * Dd) / 256, 256, 0, stream>>>(Wq, WqT);
  k_trans<<<(Dd * Dd) / 256, 256, 0, stream>>>(Wk, WkT);
  k_trans_wv<<<(Dd * Hh) / 256, 256, 0, stream>>>(Wv, WvT);

  // s = tanh(q@Wq + m@Wk + bq + bk)  (the 137-GFLOP stage, bf16 WMMA)
  k_gemm1<<<dim3(Dd / 128, Mm / 128), 256, 0, stream>>>(qb16, mb16, WqT, WkT,
                                                        bq, bk, s16);
  // logits = s @ Wv + bv
  k_gemm2<<<(Mm / 16) / 8, 256, 0, stream>>>(s16, WvT, bv, logits);
  // softmax + weighted sum of memory head-slices
  k_softmax<<<Bb * Hh, 256, 0, stream>>>(logits, bias, memory, xbuf);
  // out = x @ Wo + bo
  k_out<<<dim3(Dd / 256, Bb), 256, 0, stream>>>(xbuf, Wo, bo, out);
}
